// ProteinDiffusionModel_54872502174379
// MI455X (gfx1250) — matrix operand order repaired and checked by reading, not compiled, not run on hardware
//
#include <hip/hip_runtime.h>

typedef __attribute__((ext_vector_type(16))) _Float16 v16h;
typedef __attribute__((ext_vector_type(8)))  float    v8f;

#define PI_F 3.14159265358979323846f

static constexpr int B_ = 2, N_ = 256, A_ = 37, H_ = 128, L_ = 4;
static constexpr int KPAD_ = 416;   // 387 padded to 13*32
static constexpr int KC1_  = 13;
static constexpr int JSPLIT = 4;    // j-range split per i-tile (occupancy)

// ---------------- workspace layout (bytes) ----------------
static constexpr size_t OFF_NODEA = 0;                          // 2*256*128*4
static constexpr size_t OFF_NODEB = OFF_NODEA + 262144;
static constexpr size_t OFF_AGG   = OFF_NODEB + 262144;         // 4 partials
static constexpr size_t OFF_CM    = OFF_AGG   + 4 * 262144;
static constexpr size_t OFF_CUPD  = OFF_CM    + 6144;           // 4 partials
static constexpr size_t OFF_COORD = OFF_CUPD  + 4 * 6144;
static constexpr size_t OFF_CA    = OFF_COORD + 227328;
static constexpr size_t OFF_EW1P  = OFF_CA    + 256;
static constexpr size_t OFF_EW2P  = OFF_EW1P  + 425984;
static constexpr size_t OFF_CW1P  = OFF_EW2P  + 131072;
static constexpr size_t OFF_NW1P  = OFF_CW1P  + 131072;
static constexpr size_t OFF_NW2P  = OFF_NW1P  + 262144;

// per-layer strides in half elements of the packed weights
static constexpr int STR_EW1 = 13 * 8 * 32 * 16;   // 53248
static constexpr int STR_K4  =  4 * 8 * 32 * 16;   // 16384
static constexpr int STR_NW1 =  8 * 8 * 32 * 16;   // 32768

static constexpr int AGGP_STR  = B_ * N_ * H_;     // floats per partial
static constexpr int CUPDP_STR = B_ * N_ * 3;

__device__ __forceinline__ float silu_f(float x) {
  return x * __builtin_amdgcn_rcpf(1.0f + __expf(-x));
}

// A-fragment (16x32 f16): lane M = lane&15; halves 0..7 -> K=base..base+7,
// halves 8..15 -> K=base+16..base+23, base = kc*32 + (lane>>4)*8.
#define LOAD_A(dst, rowptr, kc)                                               \
  {                                                                           \
    const _Float16* ap_ = (rowptr) + ((kc) << 5);                             \
    _Pragma("unroll") for (int t_ = 0; t_ < 8; ++t_) {                        \
      dst[t_] = ap_[t_];                                                      \
      dst[t_ + 8] = ap_[t_ + 16];                                             \
    }                                                                         \
  }

// B-fragment: 16 contiguous halves per lane from the packed layout
#define LOAD_B(dst, wptr, kc)                                                 \
  {                                                                           \
    const _Float16* bp_ = (wptr) + (((((kc) << 3) + wave) << 5) + lane) * 16; \
    _Pragma("unroll") for (int t_ = 0; t_ < 16; ++t_) dst[t_] = bp_[t_];      \
  }

// ---------------- weight packing ----------------
__global__ void __launch_bounds__(256)
pack_kernel(const float* __restrict__ W, _Float16* __restrict__ dst,
            int K, int kcCount, int total) {
  int e = blockIdx.x * 256 + threadIdx.x;
  if (e >= total) return;
  int h    = e & 15;
  int lane = (e >> 4) & 31;
  int nt   = (e >> 9) & 7;
  int rest = e >> 12;              // l*kcCount + kc
  int kc   = rest % kcCount;
  int l    = rest / kcCount;
  int n = (nt << 4) + (lane & 15);
  int k = (kc << 5) + ((lane >> 4) << 3) + (h & 7) + ((h >> 3) << 4);
  float v = (k < K) ? W[(l * K + k) * 128 + n] : 0.0f;
  dst[e] = (_Float16)v;
}

// ---------------- node0 = res_emb + time_emb ----------------
__global__ void __launch_bounds__(256)
node0_kernel(const int* __restrict__ resIdx, const int* __restrict__ times,
             float* __restrict__ node0) {
  int idx = blockIdx.x * 256 + threadIdx.x;         // over B*N*H
  if (idx >= B_ * N_ * H_) return;
  int b = idx >> 15;
  int r = (idx >> 7) & 255;
  int h = idx & 127;
  int k = h & 63;
  float xr  = (float)resIdx[b * N_ + r];
  float ar  = xr * PI_F * exp2f(-(float)k * 0.125f);
  float vr  = (h < 64) ? __sinf(ar) : __cosf(ar);
  float xt  = (float)times[b] * (1.0f / 1000.0f);
  float at  = xt * PI_F * exp2f(-(float)k * (13.287712379549449f / 64.0f));
  float vt  = (h < 64) ? __sinf(at) : __cosf(at);
  node0[idx] = vr + vt;
}

// ---------------- per-layer atom-mean of coords ----------------
__global__ void __launch_bounds__(256)
cm_kernel(const float* __restrict__ coords, float* __restrict__ cm) {
  int idx = blockIdx.x * 256 + threadIdx.x;          // over B*N
  if (idx >= B_ * N_) return;
  float sx = 0.f, sy = 0.f, sz = 0.f;
  const float* p = coords + (size_t)idx * A_ * 3;
  for (int a = 0; a < A_; ++a) { sx += p[a*3]; sy += p[a*3+1]; sz += p[a*3+2]; }
  const float inv = 1.0f / (float)A_;
  cm[idx*3]   = sx * inv;
  cm[idx*3+1] = sy * inv;
  cm[idx*3+2] = sz * inv;
}

// ---------------- fused edge kernel ----------------
// Block = (b, 16-row i-tile, j-quadrant). Loops m over il(16) x jt(4):
//   async-copy node_j rows (contiguous 8KB) into LDS, build 16x416 ef tile,
//   WMMA chains: m = silu(silu(ef@W1)@W2), cu = silu(m@CW1)@cw2 (+biases),
//   agg/cupd partial sums per quadrant (no atomics across blocks).
__global__ void __launch_bounds__(256)
egnn_edge_kernel(const float* __restrict__ node,
                 const int*   __restrict__ resIdx,
                 const float* __restrict__ cmG,
                 const _Float16* __restrict__ w1f,
                 const _Float16* __restrict__ w2f,
                 const _Float16* __restrict__ cw1f,
                 const float* __restrict__ eb1,
                 const float* __restrict__ eb2,
                 const float* __restrict__ cb1,
                 const float* __restrict__ cw2,
                 const float* __restrict__ cb2,
                 float* __restrict__ aggP,
                 float* __restrict__ cupdP) {
  __shared__ _Float16 sEF[16 * KPAD_];
  __shared__ _Float16 sH1[16 * 128];
  __shared__ _Float16 sM [16 * 128];
  __shared__ _Float16 sNI[16 * 128];
  __shared__ float sNJ[16 * 128];     // async-filled f32 node_j rows
  __shared__ float sCM[N_ * 3];
  __shared__ int   sRES[N_];
  __shared__ float sCU[16];
  __shared__ float sCUPD[4];
  __shared__ float sAGG[8 * 32];

  const int tid  = threadIdx.x;
  const int wave = tid >> 5;
  const int lane = tid & 31;
  const int lo   = lane & 15;
  const int hi   = lane >> 4;
  const int jq   = blockIdx.x & 3;
  const int i0   = ((blockIdx.x >> 2) & 15) << 4;
  const int b    = blockIdx.x >> 6;

  for (int u = tid; u < N_ * 3; u += 256) sCM[u] = cmG[b * N_ * 3 + u];
  for (int u = tid; u < N_;     u += 256) sRES[u] = resIdx[b * N_ + u];
  for (int u = tid; u < 16 * 128; u += 256) {
    int r = u >> 7, c = u & 127;
    sNI[u] = (_Float16)node[((b * N_) + i0 + r) * H_ + c];
  }
  if (tid < 4) sCUPD[tid] = 0.0f;
  if (lane == 0) { __builtin_prefetch(w2f, 0, 0); __builtin_prefetch(cw1f, 0, 0); }

  const int   ncol  = (wave << 4) + lo;
  const float bias1 = eb1[ncol];
  const float bias2 = eb2[ncol];
  const float bias3 = cb1[ncol];
  const float cw2v  = cw2[ncol];
  const float cb2v  = cb2[0];
  float aggAcc = 0.0f;

  __syncthreads();

  for (int m = 0; m < 64; ++m) {
    const int il = m >> 2;
    const int jt = (jq << 2) + (m & 3);
    const int i  = i0 + il;
    const int j0 = jt << 4;

    // ---- kick off async global->LDS copy of the 16 node_j rows (8KB) ----
    {
      const float* gsrc = node + (size_t)(((b * N_) + j0) << 7);
      const unsigned ldsbase = (unsigned)(uintptr_t)(&sNJ[0]);  // low 32b = LDS offset
      const unsigned o0 = (unsigned)tid * 32u;
      asm volatile("global_load_async_to_lds_b128 %0, %1, %2\n\t"
                   "global_load_async_to_lds_b128 %3, %4, %2"
                   :
                   : "v"(ldsbase + o0), "v"(o0), "s"(gsrc),
                     "v"(ldsbase + o0 + 16u), "v"(o0 + 16u)
                   : "memory");
    }

    if (tid < 16) sCU[tid] = cb2v;
    // ---- build ef tile, skipping the node_j strip (filled after async) ----
    for (int u = tid; u < 16 * KPAD_; u += 256) {
      const int r = u / KPAD_;
      const int f = u - r * KPAD_;
      const int j = j0 + r;
      if (f >= 128 && f < 256) continue;       // node_j strip: async path
      float v;
      if (f < 128) {
        v = (float)sNI[il * 128 + f];
      } else if (f < 384) {
        const int k  = f - 256;
        const int kk = k & 63;
        const float x   = (float)(sRES[i] - sRES[j]);
        const float ang = x * PI_F * exp2f(-(float)kk * 0.125f);
        v = (k < 64) ? __sinf(ang) : __cosf(ang);
      } else if (f < 387) {
        const int ax = f - 384;
        v = sCM[i * 3 + ax] - sCM[j * 3 + ax];
      } else {
        v = 0.0f;
      }
      sEF[u] = (_Float16)v;
    }
    asm volatile("s_wait_asynccnt 0x0" ::: "memory");
    __syncthreads();
    // convert async-landed node_j rows f32 -> f16 into the ef strip
    for (int u = tid; u < 16 * 128; u += 256) {
      const int r = u >> 7, c = u & 127;
      sEF[r * KPAD_ + 128 + c] = (_Float16)sNJ[u];
    }
    __syncthreads();

    const _Float16* aRow1 = &sEF[lo * KPAD_ + (hi << 3)];
    // ---- GEMM1: ef @ W1, K = 416 ----
    v8f c1;
#pragma unroll
    for (int r = 0; r < 8; ++r) c1[r] = bias1;
#pragma unroll
    for (int kc = 0; kc < KC1_; ++kc) {
      v16h a, w;
      LOAD_A(a, aRow1, kc);
      LOAD_B(w, w1f, kc);
      c1 = __builtin_amdgcn_wmma_f32_16x16x32_f16(false, a, false, w, (short)0, c1, false, false);
    }
#pragma unroll
    for (int r = 0; r < 8; ++r)
      sH1[((r + (hi << 3)) << 7) + ncol] = (_Float16)silu_f(c1[r]);
    __syncthreads();

    const _Float16* aRow2 = &sH1[(lo << 7) + (hi << 3)];
    // ---- GEMM2: h1 @ W2, K = 128 ----
    v8f c2;
#pragma unroll
    for (int r = 0; r < 8; ++r) c2[r] = bias2;
#pragma unroll
    for (int kc = 0; kc < 4; ++kc) {
      v16h a, w;
      LOAD_A(a, aRow2, kc);
      LOAD_B(w, w2f, kc);
      c2 = __builtin_amdgcn_wmma_f32_16x16x32_f16(false, a, false, w, (short)0, c2, false, false);
    }
#pragma unroll
    for (int r = 0; r < 8; ++r) {
      const float s = silu_f(c2[r]);
      aggAcc += s;                                  // sum over j rows, fixed col
      sM[((r + (hi << 3)) << 7) + ncol] = (_Float16)s;
    }
    __syncthreads();

    const _Float16* aRow3 = &sM[(lo << 7) + (hi << 3)];
    // ---- GEMM3: m @ CW1, K = 128; dot with cw2 -> cu[edge] ----
    v8f c3;
#pragma unroll
    for (int r = 0; r < 8; ++r) c3[r] = bias3;
#pragma unroll
    for (int kc = 0; kc < 4; ++kc) {
      v16h a, w;
      LOAD_A(a, aRow3, kc);
      LOAD_B(w, cw1f, kc);
      c3 = __builtin_amdgcn_wmma_f32_16x16x32_f16(false, a, false, w, (short)0, c3, false, false);
    }
    float part[8];
#pragma unroll
    for (int r = 0; r < 8; ++r) part[r] = silu_f(c3[r]) * cw2v;
#pragma unroll
    for (int msk = 1; msk < 16; msk <<= 1) {
#pragma unroll
      for (int r = 0; r < 8; ++r) part[r] += __shfl_xor(part[r], msk, 32);
    }
    if (lo == 0) {
#pragma unroll
      for (int r = 0; r < 8; ++r) atomicAdd(&sCU[r + (hi << 3)], part[r]);
    }
    __syncthreads();

    // ---- cupd accumulation: sum_j cu * dirv ----
    if (tid < 16) {
      const int j = j0 + tid;
      const float cu = sCU[tid];
#pragma unroll
      for (int ax = 0; ax < 3; ++ax) {
        const float d = sCM[i * 3 + ax] - sCM[j * 3 + ax];
        atomicAdd(&sCUPD[ax], cu * d);
      }
    }

    if ((m & 3) == 3) {        // this block's j-range for row i done: flush
      __syncthreads();
      sAGG[(wave << 5) + lane] = aggAcc;
      __syncthreads();
      if (tid < 128) {
        const int w8 = tid >> 4, l2 = tid & 15;
        aggP[(size_t)jq * AGGP_STR + ((b * N_) + i) * H_ + (w8 << 4) + l2] =
            sAGG[(w8 << 5) + l2] + sAGG[(w8 << 5) + 16 + l2];
      }
      if (tid < 3) {
        cupdP[(size_t)jq * CUPDP_STR + ((b * N_) + i) * 3 + tid] = sCUPD[tid];
        sCUPD[tid] = 0.0f;
      }
      aggAcc = 0.0f;
    }
    __syncthreads();
  }
}

// ---------------- node MLP + LayerNorm ----------------
__global__ void __launch_bounds__(256)
egnn_node_kernel(const float* __restrict__ node,
                 const float* __restrict__ aggP,
                 const _Float16* __restrict__ nw1f,
                 const _Float16* __restrict__ nw2f,
                 const float* __restrict__ nb1,
                 const float* __restrict__ nb2,
                 const float* __restrict__ lng,
                 const float* __restrict__ lnb,
                 float* __restrict__ nodeOut) {
  __shared__ _Float16 sA[16 * 256];
  __shared__ _Float16 sH[16 * 128];
  __shared__ float sSum[16], sSq[16];
  const int tid  = threadIdx.x;
  const int wave = tid >> 5, lane = tid & 31, lo = lane & 15, hi = lane >> 4;
  const int b  = blockIdx.x >> 4;
  const int n0 = (blockIdx.x & 15) << 4;

  for (int u = tid; u < 16 * 256; u += 256) {
    const int r = u >> 8, c = u & 255;
    float v;
    if (c < 128) {
      v = node[((b * N_) + n0 + r) * H_ + c];
    } else {
      const int o = ((b * N_) + n0 + r) * H_ + (c - 128);
      v = aggP[o] + aggP[AGGP_STR + o] + aggP[2 * AGGP_STR + o] + aggP[3 * AGGP_STR + o];
    }
    sA[u] = (_Float16)v;
  }
  if (tid < 16) { sSum[tid] = 0.0f; sSq[tid] = 0.0f; }
  __syncthreads();

  const int ncol = (wave << 4) + lo;
  const _Float16* aRow1 = &sA[(lo << 8) + (hi << 3)];
  v8f c1;
#pragma unroll
  for (int r = 0; r < 8; ++r) c1[r] = nb1[ncol];
#pragma unroll
  for (int kc = 0; kc < 8; ++kc) {
    v16h a, w;
    LOAD_A(a, aRow1, kc);
    LOAD_B(w, nw1f, kc);
    c1 = __builtin_amdgcn_wmma_f32_16x16x32_f16(false, a, false, w, (short)0, c1, false, false);
  }
#pragma unroll
  for (int r = 0; r < 8; ++r)
    sH[((r + (hi << 3)) << 7) + ncol] = (_Float16)silu_f(c1[r]);
  __syncthreads();

  const _Float16* aRow2 = &sH[(lo << 7) + (hi << 3)];
  v8f c2;
#pragma unroll
  for (int r = 0; r < 8; ++r) c2[r] = nb2[ncol];
#pragma unroll
  for (int kc = 0; kc < 4; ++kc) {
    v16h a, w;
    LOAD_A(a, aRow2, kc);
    LOAD_B(w, nw2f, kc);
    c2 = __builtin_amdgcn_wmma_f32_16x16x32_f16(false, a, false, w, (short)0, c2, false, false);
  }
#pragma unroll
  for (int r = 0; r < 8; ++r) {
    atomicAdd(&sSum[r + (hi << 3)], c2[r]);
    atomicAdd(&sSq [r + (hi << 3)], c2[r] * c2[r]);
  }
  __syncthreads();
  const float g = lng[ncol], be = lnb[ncol];
#pragma unroll
  for (int r = 0; r < 8; ++r) {
    const int row = r + (hi << 3);
    const float mu  = sSum[row] * (1.0f / 128.0f);
    const float var = sSq[row] * (1.0f / 128.0f) - mu * mu;
    nodeOut[((b * N_) + n0 + row) * H_ + ncol] =
        (c2[r] - mu) * rsqrtf(var + 1e-5f) * g + be;
  }
}

// ---------------- coordinate update ----------------
__global__ void __launch_bounds__(256)
coord_update_kernel(const float* __restrict__ cupdP,
                    const float* __restrict__ mask,
                    float* __restrict__ coords) {
  int idx = blockIdx.x * 256 + threadIdx.x;          // over B*N*A
  if (idx >= B_ * N_ * A_) return;
  int bn = idx / A_;
  float m = mask[idx];
#pragma unroll
  for (int ax = 0; ax < 3; ++ax) {
    const int o = bn * 3 + ax;
    const float cu = cupdP[o] + cupdP[CUPDP_STR + o] +
                     cupdP[2 * CUPDP_STR + o] + cupdP[3 * CUPDP_STR + o];
    coords[idx * 3 + ax] += cu * m;
  }
}

// ---------------- diffusion scale ----------------
__global__ void ca_kernel(const int* __restrict__ times, float* __restrict__ ca) {
  int b = threadIdx.x;
  if (b >= B_) return;
  float tn  = (float)times[b] * (1.0f / 1000.0f);
  int   idx = (int)tn;                                // faithful: floor(t/1000)
  float prod = 1.0f;
  for (int k = 0; k <= idx; ++k) {
    float beta = 1e-4f + (2e-2f - 1e-4f) * (float)k / 999.0f;
    prod *= (1.0f - beta);
  }
  ca[b] = prod;
}

__global__ void __launch_bounds__(256)
final_kernel(const float* __restrict__ coords0,
             const float* __restrict__ cur,
             const float* __restrict__ mask,
             const float* __restrict__ ca,
             float* __restrict__ out) {
  int idx = blockIdx.x * 256 + threadIdx.x;           // over B*N*A*3
  if (idx >= B_ * N_ * A_ * 3) return;
  int b   = idx / (N_ * A_ * 3);
  int bna = idx / 3;
  float a  = ca[b];
  float s1 = sqrtf(a);
  float s2 = sqrtf(1.0f - a);
  out[idx] = (coords0[idx] - s1 * cur[idx]) / s2 * mask[bna];
}

// ---------------- host launcher ----------------
extern "C" void kernel_launch(void* const* d_in, const int* in_sizes, int n_in,
                              void* d_out, int out_size, void* d_ws, size_t ws_size,
                              hipStream_t stream) {
  (void)in_sizes; (void)n_in; (void)out_size; (void)ws_size;
  const float* coords   = (const float*)d_in[0];
  const int*   resIdx   = (const int*)d_in[1];
  const int*   times    = (const int*)d_in[2];
  const float* mask     = (const float*)d_in[3];
  const float* edge_w1  = (const float*)d_in[4];
  const float* edge_b1  = (const float*)d_in[5];
  const float* edge_w2  = (const float*)d_in[6];
  const float* edge_b2  = (const float*)d_in[7];
  const float* node_w1  = (const float*)d_in[8];
  const float* node_b1  = (const float*)d_in[9];
  const float* node_w2  = (const float*)d_in[10];
  const float* node_b2  = (const float*)d_in[11];
  const float* ln_g     = (const float*)d_in[12];
  const float* ln_b     = (const float*)d_in[13];
  const float* coord_w1 = (const float*)d_in[14];
  const float* coord_b1 = (const float*)d_in[15];
  const float* coord_w2 = (const float*)d_in[16];
  const float* coord_b2 = (const float*)d_in[17];

  char* ws = (char*)d_ws;
  float*    nodeA = (float*)(ws + OFF_NODEA);
  float*    nodeB = (float*)(ws + OFF_NODEB);
  float*    aggP  = (float*)(ws + OFF_AGG);
  float*    cm    = (float*)(ws + OFF_CM);
  float*    cupdP = (float*)(ws + OFF_CUPD);
  float*    ccur  = (float*)(ws + OFF_COORD);
  float*    ca    = (float*)(ws + OFF_CA);
  _Float16* ew1p  = (_Float16*)(ws + OFF_EW1P);
  _Float16* ew2p  = (_Float16*)(ws + OFF_EW2P);
  _Float16* cw1p  = (_Float16*)(ws + OFF_CW1P);
  _Float16* nw1p  = (_Float16*)(ws + OFF_NW1P);
  _Float16* nw2p  = (_Float16*)(ws + OFF_NW2P);

  // pack weights into WMMA B-fragment order (f32 -> f16)
  pack_kernel<<<(L_*13*4096 + 255)/256, 256, 0, stream>>>(edge_w1,  ew1p, 387, 13, L_*13*4096);
  pack_kernel<<<(L_*4 *4096 + 255)/256, 256, 0, stream>>>(edge_w2,  ew2p, 128,  4, L_*4*4096);
  pack_kernel<<<(L_*4 *4096 + 255)/256, 256, 0, stream>>>(coord_w1, cw1p, 128,  4, L_*4*4096);
  pack_kernel<<<(L_*8 *4096 + 255)/256, 256, 0, stream>>>(node_w1,  nw1p, 256,  8, L_*8*4096);
  pack_kernel<<<(L_*4 *4096 + 255)/256, 256, 0, stream>>>(node_w2,  nw2p, 128,  4, L_*4*4096);

  node0_kernel<<<(B_*N_*H_ + 255)/256, 256, 0, stream>>>(resIdx, times, nodeA);
  hipMemcpyAsync(ccur, coords, (size_t)B_*N_*A_*3*sizeof(float),
                 hipMemcpyDeviceToDevice, stream);

  float* nIn = nodeA;
  float* nOut = nodeB;
  for (int l = 0; l < L_; ++l) {
    cm_kernel<<<2, 256, 0, stream>>>(ccur, cm);
    egnn_edge_kernel<<<B_*(N_/16)*JSPLIT, 256, 0, stream>>>(
        nIn, resIdx, cm,
        ew1p + (size_t)l * STR_EW1,
        ew2p + (size_t)l * STR_K4,
        cw1p + (size_t)l * STR_K4,
        edge_b1 + l*H_, edge_b2 + l*H_,
        coord_b1 + l*H_, coord_w2 + l*H_, coord_b2 + l,
        aggP, cupdP);
    egnn_node_kernel<<<B_*(N_/16), 256, 0, stream>>>(
        nIn, aggP,
        nw1p + (size_t)l * STR_NW1,
        nw2p + (size_t)l * STR_K4,
        node_b1 + l*H_, node_b2 + l*H_,
        ln_g + l*H_, ln_b + l*H_,
        nOut);
    coord_update_kernel<<<(B_*N_*A_ + 255)/256, 256, 0, stream>>>(cupdP, mask, ccur);
    float* t = nIn; nIn = nOut; nOut = t;
  }

  ca_kernel<<<1, 32, 0, stream>>>(times, ca);
  final_kernel<<<(B_*N_*A_*3 + 255)/256, 256, 0, stream>>>(
      coords, ccur, mask, ca, (float*)d_out);
}
